// TokenLevelSSMEncoder_83958020702554
// MI455X (gfx1250) — compile-verified, matrix operand order checked by reading, not run on hardware
//
#include <hip/hip_runtime.h>

// ---------------------------------------------------------------------------
// TokenLevelSSMEncoder fused kernel for MI455X (gfx1250, wave32, WMMA)
// One workgroup (256 threads = 8 wave32) per batch element.
// All GEMMs use v_wmma_f32_16x16x32_f16 (f16 operands, f32 accumulate).
// Weights are pre-packed (f32 -> f16, WMMA B-fragment order) into d_ws by a
// small one-shot kernel, so the hot loop issues pure b128 loads + WMMA.
// ---------------------------------------------------------------------------

typedef __attribute__((ext_vector_type(16))) _Float16 v16h;
typedef __attribute__((ext_vector_type(8)))  float    v8f;

#define T_      64
#define D_      512
#define S_      64
#define DEPTH_  2
#define LN_EPS  1e-5f

// Padded LDS row strides (avoid 64-bank conflicts)
#define HS   520   // hidden tile row stride in halves
#define DS_  65    // decay/drive row stride in floats
#define HSS  72    // hs tile row stride in halves

// LDS layout (bytes)
#define OFF_H      0                         // _Float16 [64][HS]   -> 66560
#define OFF_DECAY  66560                     // float    [64][DS_]  -> 16640
#define OFF_DRIVE  (OFF_DECAY + 16640)       // float    [64][DS_]  -> 16640
#define OFF_HS     (OFF_DRIVE + 16640)       // _Float16 [64][HSS]  -> 9216
#define OFF_HT     (OFF_HS + 9216)           // float    [64]       -> 256
#define OFF_SHOT   (OFF_HT + 256)            // float    [512]      -> 2048
#define OFF_POOL   (OFF_SHOT + 2048)         // float    [512]      -> 2048
#define OFF_RED    (OFF_POOL + 2048)         // float    [256]      -> 1024
#define OFF_RED2   (OFF_RED + 1024)          // float    [256]      -> 1024
#define SMEM_BYTES (OFF_RED2 + 1024)         // = 115456

// Packed-weight layout in d_ws: 6 matrices (li*3 + {Wd,Wb,Wc}) of 2048
// (frag,lane) chunks x 16 halves each.
#define PK_MAT     32768                     // halves per packed matrix
#define WS_NEEDED  (6u * PK_MAT * 2u)        // 393216 bytes

// ---------------------------------------------------------------------------
// One-shot weight packer: f32 -> f16 in WMMA B-fragment order.
//   Wd/Wb (K=512,N=64):  frag = kk*4 + nt   (kk 0..15, nt 0..3)
//   Wc    (K=64,N=512):  frag = kk*32 + nb  (kk 0..1,  nb 0..31)
// Lane's 16 halves: j<8 -> K=kbase+j ; j>=8 -> K=kbase+16+(j-8),
// kbase = kk*32 + (lane&16 ? 8 : 0), col n = blk*16 + (lane&15).
// ---------------------------------------------------------------------------
__global__ void __launch_bounds__(256)
pack_weights_kernel(const float* __restrict__ Wd,
                    const float* __restrict__ Wb,
                    const float* __restrict__ Wc,
                    _Float16* __restrict__ out)
{
    const int gid  = blockIdx.x * blockDim.x + threadIdx.x;   // 0..12287
    const int mat  = gid >> 11;          // 0..5 = li*3 + {0:Wd,1:Wb,2:Wc}
    const int fl   = gid & 2047;         // frag*32 + lane
    const int lane = fl & 31;
    const int frag = fl >> 5;
    const int li    = mat / 3;
    const int which = mat % 3;

    const float* src;
    int ldn, kbase, n;
    if (which == 2) {                    // Wc : K=64, N=512
        src = Wc + (size_t)li * S_ * D_;
        ldn = D_;
        const int kk = frag >> 5;        // 0..1
        const int nb = frag & 31;        // 0..31
        kbase = kk * 32 + ((lane & 16) ? 8 : 0);
        n     = nb * 16 + (lane & 15);
    } else {                             // Wd/Wb : K=512, N=64
        src = (which == 0 ? Wd : Wb) + (size_t)li * D_ * S_;
        ldn = S_;
        const int kk = frag >> 2;        // 0..15
        const int nt = frag & 3;         // 0..3
        kbase = kk * 32 + ((lane & 16) ? 8 : 0);
        n     = nt * 16 + (lane & 15);
    }

    v16h v;
    #pragma unroll
    for (int j = 0; j < 8; ++j) {
        v[j]     = (_Float16)src[(size_t)(kbase + j) * ldn + n];
        v[8 + j] = (_Float16)src[(size_t)(kbase + 16 + j) * ldn + n];
    }
    *(v16h*)(out + (size_t)mat * PK_MAT + (size_t)fl * 16) = v;
}

// ---------------------------------------------------------------------------
// Main fused kernel. PACKED selects b128 fragment loads from d_ws vs the
// scalar f32 gather fallback (used only if ws_size is too small).
// ---------------------------------------------------------------------------
template <bool PACKED>
__global__ void __launch_bounds__(256, 1)
ssm_encoder_kernel(const float* __restrict__ tokens,
                   const float* __restrict__ pos_emb,
                   const float* __restrict__ A_log,
                   const float* __restrict__ Wd, const float* __restrict__ bd,
                   const float* __restrict__ Wb, const float* __restrict__ bb,
                   const float* __restrict__ Wc, const float* __restrict__ bc,
                   const float* __restrict__ Dp,
                   const float* __restrict__ Ws, const float* __restrict__ bs,
                   const float* __restrict__ gamma, const float* __restrict__ beta,
                   const _Float16* __restrict__ packW,
                   float* __restrict__ out_hidden, float* __restrict__ out_shot)
{
    extern __shared__ char smem[];
    _Float16* sm_h     = (_Float16*)(smem + OFF_H);
    float*    sm_decay = (float*)   (smem + OFF_DECAY);
    float*    sm_drive = (float*)   (smem + OFF_DRIVE);
    _Float16* sm_hs    = (_Float16*)(smem + OFF_HS);
    float*    sm_hT    = (float*)   (smem + OFF_HT);
    float*    sm_shot  = (float*)   (smem + OFF_SHOT);
    float*    sm_pool  = (float*)   (smem + OFF_POOL);
    float*    sm_red   = (float*)   (smem + OFF_RED);
    float*    sm_red2  = (float*)   (smem + OFF_RED2);

    const int tid  = threadIdx.x;
    const int lane = tid & 31;
    const int wave = tid >> 5;
    const int b    = blockIdx.x;
    const long base_td = (long)b * T_ * D_;

    // ---- Stage hidden = tokens[b] + pos_emb into LDS as f16 -----------------
    {
        const float4* tok4 = (const float4*)(tokens + base_td);
        const float4* pe4  = (const float4*)pos_emb;
        for (int i4 = tid; i4 < T_ * D_ / 4; i4 += 256) {
            float4 a = tok4[i4];
            float4 p = pe4[i4];
            int n = i4 * 4;
            int t = n >> 9;
            int d = n & 511;
            _Float16* dst = sm_h + t * HS + d;
            dst[0] = (_Float16)(a.x + p.x);
            dst[1] = (_Float16)(a.y + p.y);
            dst[2] = (_Float16)(a.z + p.z);
            dst[3] = (_Float16)(a.w + p.w);
        }
    }
    __syncthreads();

    const int r     = lane & 15;           // row/col within 16x16 tile
    const int khalf = (lane & 16) ? 8 : 0; // A/B fragment K sub-offset
    const int thalf = (lane & 16) ? 8 : 0; // C fragment M sub-offset

    for (int li = 0; li < DEPTH_; ++li) {
        const float* wWd = Wd + (size_t)li * D_ * S_;
        const float* wWb = Wb + (size_t)li * D_ * S_;
        const float* wbd = bd + li * S_;
        const float* wbb = bb + li * S_;
        const float* wA  = A_log + li * S_;
        const float* wWc = Wc + (size_t)li * S_ * D_;
        const float* wbc = bc + li * D_;
        const float* wDp = Dp + li * D_;
        const float* wWs = Ws + (size_t)li * S_ * D_;
        const float* wbs = bs + li * D_;
        const _Float16* pkWd = packW + (size_t)(li * 3 + 0) * PK_MAT;
        const _Float16* pkWb = packW + (size_t)(li * 3 + 1) * PK_MAT;
        const _Float16* pkWc = packW + (size_t)(li * 3 + 2) * PK_MAT;

        // ---- GEMM1: P = hidden@Wd, Q = hidden@Wb (both [64x64]) ------------
        // wave w: fixed nt = w&3 (B-fragments shared), mt in {w>>2, w>>2+2}
        {
            const int nt   = wave & 3;
            const int mt0  = wave >> 2;       // 0..1
            const int mt1  = mt0 + 2;
            const int ncol = nt * 16 + r;     // s column
            v8f accP0 = {}, accP1 = {}, accQ0 = {}, accQ1 = {};

            for (int kk = 0; kk < 16; ++kk) {
                const int k0 = kk * 32 + khalf;
                v16h a0, a1;
                {
                    const _Float16* p0 = sm_h + (mt0 * 16 + r) * HS + k0;
                    const _Float16* p1 = sm_h + (mt1 * 16 + r) * HS + k0;
                    #pragma unroll
                    for (int j = 0; j < 8; ++j) {
                        a0[j] = p0[j];  a0[8 + j] = p0[16 + j];
                        a1[j] = p1[j];  a1[8 + j] = p1[16 + j];
                    }
                }
                v16h bf0, bf1;
                if (PACKED) {
                    const size_t fo = ((size_t)(kk * 4 + nt) * 32 + lane) * 16;
                    bf0 = *(const v16h*)(pkWd + fo);
                    bf1 = *(const v16h*)(pkWb + fo);
                } else {
                    const float* pd = wWd + (size_t)k0 * S_ + ncol;
                    const float* pb = wWb + (size_t)k0 * S_ + ncol;
                    #pragma unroll
                    for (int j = 0; j < 8; ++j) {
                        bf0[j]     = (_Float16)pd[j * S_];
                        bf0[8 + j] = (_Float16)pd[(16 + j) * S_];
                        bf1[j]     = (_Float16)pb[j * S_];
                        bf1[8 + j] = (_Float16)pb[(16 + j) * S_];
                    }
                }
                accP0 = __builtin_amdgcn_wmma_f32_16x16x32_f16(false, a0, false, bf0, (short)0, accP0, false, false);
                accP1 = __builtin_amdgcn_wmma_f32_16x16x32_f16(false, a1, false, bf0, (short)0, accP1, false, false);
                accQ0 = __builtin_amdgcn_wmma_f32_16x16x32_f16(false, a0, false, bf1, (short)0, accQ0, false, false);
                accQ1 = __builtin_amdgcn_wmma_f32_16x16x32_f16(false, a1, false, bf1, (short)0, accQ1, false, false);
            }
            // epilogue: delta = softplus(P + bd) -> sm_decay ; (Q + bb) -> sm_drive
            const float bdv = wbd[ncol];
            const float bbv = wbb[ncol];
            #pragma unroll
            for (int rr = 0; rr < 8; ++rr) {
                int t0 = mt0 * 16 + rr + thalf;
                int t1 = mt1 * 16 + rr + thalf;
                float p0 = accP0[rr] + bdv;
                float p1 = accP1[rr] + bdv;
                p0 = (p0 > 20.f) ? p0 : __logf(1.f + __expf(p0));
                p1 = (p1 > 20.f) ? p1 : __logf(1.f + __expf(p1));
                sm_decay[t0 * DS_ + ncol] = p0;               // delta (temp)
                sm_decay[t1 * DS_ + ncol] = p1;
                sm_drive[t0 * DS_ + ncol] = accQ0[rr] + bbv;  // B-proj (temp)
                sm_drive[t1 * DS_ + ncol] = accQ1[rr] + bbv;
            }
        }
        __syncthreads();

        // ---- elementwise: decay = exp(-delta*exp(A)), drive = delta*Bproj --
        for (int i = tid; i < T_ * S_; i += 256) {
            int t = i >> 6, s = i & 63;
            float delta = sm_decay[t * DS_ + s];
            float eA    = __expf(wA[s]);
            sm_decay[t * DS_ + s] = __expf(-delta * eA);
            sm_drive[t * DS_ + s] = delta * sm_drive[t * DS_ + s];
        }
        __syncthreads();

        // ---- sequential scan over T (64 states in parallel) ----------------
        if (tid < S_) {
            const int s = tid;
            float h = 0.f;
            for (int t = 0; t < T_; ++t) {
                h = sm_decay[t * DS_ + s] * h + sm_drive[t * DS_ + s];
                sm_hs[t * HSS + s] = (_Float16)h;
            }
            sm_hT[s] = h;
        }
        __syncthreads();

        // ---- shotvec = hT @ Ws + bs (GEMV, VALU) ---------------------------
        {
            const int d0 = tid * 2;
            float s0 = 0.f, s1 = 0.f;
            for (int s = 0; s < S_; ++s) {
                float h = sm_hT[s];
                float2 w = *(const float2*)(wWs + s * D_ + d0);
                s0 += h * w.x;
                s1 += h * w.y;
            }
            float2 bsv = *(const float2*)(wbs + d0);
            sm_shot[d0]     = s0 + bsv.x;
            sm_shot[d0 + 1] = s1 + bsv.y;
        }
        __syncthreads();

        // ---- GEMM2: hidden' = hs@Wc + bc + hidden*Dp + shotvec -------------
        {
            const int mt = wave & 3;
            v16h a0, a1;   // hs A-fragments for K=0..31 and K=32..63
            {
                const _Float16* p0 = sm_hs + (mt * 16 + r) * HSS + khalf;
                const _Float16* p1 = p0 + 32;
                #pragma unroll
                for (int j = 0; j < 8; ++j) {
                    a0[j] = p0[j];  a0[8 + j] = p0[16 + j];
                    a1[j] = p1[j];  a1[8 + j] = p1[16 + j];
                }
            }
            for (int it = 0; it < 16; ++it) {
                const int nb = (wave >> 2) + 2 * it;   // d block 0..31
                const int d  = nb * 16 + r;
                v16h b0, b1;
                if (PACKED) {
                    b0 = *(const v16h*)(pkWc + ((size_t)(0 * 32 + nb) * 32 + lane) * 16);
                    b1 = *(const v16h*)(pkWc + ((size_t)(1 * 32 + nb) * 32 + lane) * 16);
                } else {
                    const float* pc = wWc + (size_t)khalf * D_ + d;
                    #pragma unroll
                    for (int j = 0; j < 8; ++j) {
                        b0[j]     = (_Float16)pc[j * D_];
                        b0[8 + j] = (_Float16)pc[(16 + j) * D_];
                        b1[j]     = (_Float16)pc[(32 + j) * D_];
                        b1[8 + j] = (_Float16)pc[(48 + j) * D_];
                    }
                }
                v8f acc = {};
                acc = __builtin_amdgcn_wmma_f32_16x16x32_f16(false, a0, false, b0, (short)0, acc, false, false);
                acc = __builtin_amdgcn_wmma_f32_16x16x32_f16(false, a1, false, b1, (short)0, acc, false, false);

                const float bcv = wbc[d];
                const float dpv = wDp[d];
                const float sv  = sm_shot[d];
                #pragma unroll
                for (int rr = 0; rr < 8; ++rr) {
                    int t = mt * 16 + rr + thalf;
                    float old = (float)sm_h[t * HS + d];
                    float y   = acc[rr] + bcv + old * dpv + sv;
                    sm_h[t * HS + d] = (_Float16)y;
                    if (li == DEPTH_ - 1)
                        out_hidden[base_td + t * D_ + d] = y;
                }
            }
        }
        __syncthreads();
    }

    // ---- pooled = mean_t(hidden) + shotvec ; LayerNorm -> shot -------------
    {
        const int d0 = tid * 2;
        float m0 = 0.f, m1 = 0.f;
        for (int t = 0; t < T_; ++t) {
            m0 += (float)sm_h[t * HS + d0];
            m1 += (float)sm_h[t * HS + d0 + 1];
        }
        m0 = m0 * (1.f / T_) + sm_shot[d0];
        m1 = m1 * (1.f / T_) + sm_shot[d0 + 1];
        sm_pool[d0]     = m0;
        sm_pool[d0 + 1] = m1;
        sm_red[tid]  = m0 + m1;
        sm_red2[tid] = m0 * m0 + m1 * m1;
    }
    __syncthreads();
    for (int stp = 128; stp > 0; stp >>= 1) {
        if (tid < stp) {
            sm_red[tid]  += sm_red[tid + stp];
            sm_red2[tid] += sm_red2[tid + stp];
        }
        __syncthreads();
    }
    {
        const float mu  = sm_red[0] * (1.f / D_);
        const float var = sm_red2[0] * (1.f / D_) - mu * mu;
        const float inv = rsqrtf(var + LN_EPS);
        const int d0 = tid * 2;
        float* os = out_shot + (long)b * D_;
        os[d0]     = (sm_pool[d0]     - mu) * inv * gamma[d0]     + beta[d0];
        os[d0 + 1] = (sm_pool[d0 + 1] - mu) * inv * gamma[d0 + 1] + beta[d0 + 1];
    }
}

extern "C" void kernel_launch(void* const* d_in, const int* in_sizes, int n_in,
                              void* d_out, int out_size, void* d_ws, size_t ws_size,
                              hipStream_t stream) {
    const float* tokens  = (const float*)d_in[0];
    const float* pos_emb = (const float*)d_in[1];
    const float* A_log   = (const float*)d_in[2];
    const float* Wd      = (const float*)d_in[3];
    const float* bd      = (const float*)d_in[4];
    const float* Wb      = (const float*)d_in[5];
    const float* bb      = (const float*)d_in[6];
    const float* Wc      = (const float*)d_in[7];
    const float* bc      = (const float*)d_in[8];
    const float* Dp      = (const float*)d_in[9];
    const float* Ws      = (const float*)d_in[10];
    const float* bs      = (const float*)d_in[11];
    const float* gamma   = (const float*)d_in[12];
    const float* beta    = (const float*)d_in[13];

    const int B = in_sizes[0] / (T_ * D_);
    float* out_hidden = (float*)d_out;
    float* out_shot   = out_hidden + (size_t)B * T_ * D_;

    const bool packed = (ws_size >= (size_t)WS_NEEDED);

    if (packed) {
        _Float16* packW = (_Float16*)d_ws;
        pack_weights_kernel<<<48, 256, 0, stream>>>(Wd, Wb, Wc, packW);
        (void)hipFuncSetAttribute(
            reinterpret_cast<const void*>(&ssm_encoder_kernel<true>),
            hipFuncAttributeMaxDynamicSharedMemorySize, (int)SMEM_BYTES);
        ssm_encoder_kernel<true><<<B, 256, SMEM_BYTES, stream>>>(
            tokens, pos_emb, A_log, Wd, bd, Wb, bb, Wc, bc, Dp, Ws, bs,
            gamma, beta, packW, out_hidden, out_shot);
    } else {
        (void)hipFuncSetAttribute(
            reinterpret_cast<const void*>(&ssm_encoder_kernel<false>),
            hipFuncAttributeMaxDynamicSharedMemorySize, (int)SMEM_BYTES);
        ssm_encoder_kernel<false><<<B, 256, SMEM_BYTES, stream>>>(
            tokens, pos_emb, A_log, Wd, bd, Wb, bb, Wc, bc, Dp, Ws, bs,
            gamma, beta, (const _Float16*)d_ws, out_hidden, out_shot);
    }
}
